// GPSA_3590592659849
// MI455X (gfx1250) — compile-verified, matrix operand order checked by reading, not run on hardware
//
#include <hip/hip_runtime.h>
#include <hip/hip_bf16.h>

typedef __attribute__((ext_vector_type(16))) _Float16 v16h;
typedef __attribute__((ext_vector_type(8)))  float    v8f;
typedef unsigned int u32x4 __attribute__((ext_vector_type(4)));
typedef int          i32x4 __attribute__((ext_vector_type(4)));
typedef int          i32x8 __attribute__((ext_vector_type(8)));

#define HCNT   16
#define HD     48
#define CDIM   768
#define NSEQ   576
#define BATCH  8
#define GRIDS  24
#define MROWS  (BATCH*NSEQ)   // 4608
#define KPAD   64
#define CHUNKR 96             // K-panel rows staged per TDM chunk

#define WMMA_F16(a, b, c) \
  __builtin_amdgcn_wmma_f32_16x16x32_f16(false, (a), false, (b), (short)0, (c), false, false)

// ---------------------------------------------------------------------------
// K-major per-lane f16 fragment loader for 16x16x32 WMMA (A layout; B is fed
// as N x K through the same function). Two b128 loads per lane (ds_load_b128
// when the pointer derives from LDS).
// ---------------------------------------------------------------------------
__device__ __forceinline__ v16h load_frag_f16(const _Float16* __restrict__ p,
                                              int ld, int row0, int k0) {
  const int lane = threadIdx.x & 31;
  const _Float16* rp = p + (size_t)(row0 + (lane & 15)) * ld + k0 + ((lane >> 4) << 3);
  union { v16h v; uint4 u[2]; } t;
  t.u[0] = *reinterpret_cast<const uint4*>(rp);
  t.u[1] = *reinterpret_cast<const uint4*>(rp + 16);
  return t.v;
}

// ---------------------------------------------------------------------------
// Tensor Data Mover: async 2-D tile (f16) global -> LDS.
// D# group0: count=1, lds_addr, 57-bit global addr, type=2.
// D# group1: data_size=2B, tensor dims/strides, tile dims.
// ---------------------------------------------------------------------------
__device__ __forceinline__ void tdm_load_2d_f16(unsigned lds_addr, const void* gaddr,
                                                int tile_d0, int tile_d1,
                                                int tensor_d0, int tensor_d1,
                                                int stride_d0) {
#if __has_builtin(__builtin_amdgcn_tensor_load_to_lds)
  unsigned long long ga = (unsigned long long)(uintptr_t)gaddr;
  u32x4 g0 = { 1u,                                   // count=1, is_restore=0
               lds_addr,                             // bits 63:32
               (unsigned)(ga & 0xffffffffu),         // global_addr[31:0]
               (unsigned)((ga >> 32) & 0x01ffffffu) | (2u << 30) };  // addr[56:32] | type=2
  i32x8 g1 = { (int)(1u << 16),                          // data_size=1 -> 2 bytes
               (int)((unsigned)(tensor_d0 & 0xffff) << 16),
               (int)(((unsigned)tensor_d0 >> 16) | ((unsigned)(tensor_d1 & 0xffff) << 16)),
               (int)(((unsigned)tensor_d1 >> 16) | ((unsigned)(tile_d0 & 0xffff) << 16)),
               (int)(tile_d1 & 0xffff),                  // tile_dim1 (tile_dim2=0)
               stride_d0,                                // tensor_dim0_stride[31:0]
               0, 0 };
  i32x4 gz = { 0, 0, 0, 0 };
#if __clang_major__ >= 23
  i32x8 gz8 = { 0, 0, 0, 0, 0, 0, 0, 0 };
  __builtin_amdgcn_tensor_load_to_lds(g0, g1, gz, gz, gz8, 0);
#else
  __builtin_amdgcn_tensor_load_to_lds(g0, g1, gz, gz, 0);
#endif
#else
  (void)lds_addr; (void)gaddr; (void)tile_d0; (void)tile_d1;
  (void)tensor_d0; (void)tensor_d1; (void)stride_d0;
#endif
}

template <int N>
__device__ __forceinline__ void wait_tensorcnt() {
#if __has_builtin(__builtin_amdgcn_s_wait_tensorcnt)
  __builtin_amdgcn_s_wait_tensorcnt((short)N);
#else
  asm volatile("s_wait_tensorcnt %0" :: "i"(N) : "memory");
#endif
}

// ---------------------------------------------------------------------------
// fp32 -> f16 elementwise conversion
// ---------------------------------------------------------------------------
__global__ void __launch_bounds__(256)
cvt_f32_to_f16(const float* __restrict__ s, _Float16* __restrict__ d, int n) {
  int i = blockIdx.x * blockDim.x + threadIdx.x;
  if (i < n) d[i] = (_Float16)s[i];
}

// ---------------------------------------------------------------------------
// 768-K WMMA GEMM, 32(M) x 64(N) tile per wave. K loop unrolled x2 with
// explicit ping-pong fragment sets; accumulators are individually named so
// the register allocator can pin each WMMA C/D octet (no acc copies).
// B given row-major as (n, k) = W so D = A * W^T matches nn.Linear.
// ---------------------------------------------------------------------------
#define MODE_Q    0
#define MODE_K    1
#define MODE_V    2
#define MODE_PROJ 3

__global__ void __launch_bounds__(256, 2)
gemm768_wmma(const _Float16* __restrict__ A, const _Float16* __restrict__ B,
             _Float16* __restrict__ oh, float* __restrict__ of,
             const float* __restrict__ bias, int mode, float alpha) {
  const int wave = blockIdx.x * (blockDim.x >> 5) + (threadIdx.x >> 5);
  const int mt = wave / 12;          // 144 M-tiles of 32
  const int nt = wave % 12;          // 12 N-tiles of 64
  if (mt >= MROWS / 32) return;
  const int lane = threadIdx.x & 31;
  const int m0 = mt * 32;
  const int n0 = nt * 64;

  v8f c00 = {}, c01 = {}, c02 = {}, c03 = {};   // M-subtile 0, N-subtiles 0..3
  v8f c10 = {}, c11 = {}, c12 = {}, c13 = {};   // M-subtile 1, N-subtiles 0..3
  v16h aA0, aA1, bA0, bA1, bA2, bA3;   // ping set
  v16h aB0, aB1, bB0, bB1, bB2, bB3;   // pong set

#define G_LOADSET(SUF, KK)                                  \
  a##SUF##0 = load_frag_f16(A, CDIM, m0,      (KK));        \
  a##SUF##1 = load_frag_f16(A, CDIM, m0 + 16, (KK));        \
  b##SUF##0 = load_frag_f16(B, CDIM, n0 +  0, (KK));        \
  b##SUF##1 = load_frag_f16(B, CDIM, n0 + 16, (KK));        \
  b##SUF##2 = load_frag_f16(B, CDIM, n0 + 32, (KK));        \
  b##SUF##3 = load_frag_f16(B, CDIM, n0 + 48, (KK));

#define G_WMMASET(SUF)                                      \
  c00 = WMMA_F16(a##SUF##0, b##SUF##0, c00);                \
  c10 = WMMA_F16(a##SUF##1, b##SUF##0, c10);                \
  c01 = WMMA_F16(a##SUF##0, b##SUF##1, c01);                \
  c11 = WMMA_F16(a##SUF##1, b##SUF##1, c11);                \
  c02 = WMMA_F16(a##SUF##0, b##SUF##2, c02);                \
  c12 = WMMA_F16(a##SUF##1, b##SUF##2, c12);                \
  c03 = WMMA_F16(a##SUF##0, b##SUF##3, c03);                \
  c13 = WMMA_F16(a##SUF##1, b##SUF##3, c13);

  G_LOADSET(A, 0)
  for (int k0 = 0; k0 < CDIM - 64; k0 += 64) {   // 11 iterations (24 K-steps total)
    __builtin_prefetch(A + (size_t)(m0 + (lane & 15)) * CDIM + k0 + 128, 0, 1);
    G_LOADSET(B, k0 + 32)
    G_WMMASET(A)
    G_LOADSET(A, k0 + 64)
    G_WMMASET(B)
  }
  G_LOADSET(B, CDIM - 32)
  G_WMMASET(A)
  G_WMMASET(B)
#undef G_LOADSET
#undef G_WMMASET

  const int half8 = (lane >> 4) << 3;     // D layout: M = vgpr + 8*(lane>>4)
  const int cn = lane & 15;               //           N = lane & 15
  #pragma unroll
  for (int ms = 0; ms < 2; ++ms) {
    #pragma unroll
    for (int s = 0; s < 4; ++s) {
      v8f acc = (ms == 0) ? ((s == 0) ? c00 : (s == 1) ? c01 : (s == 2) ? c02 : c03)
                          : ((s == 0) ? c10 : (s == 1) ? c11 : (s == 2) ? c12 : c13);
      const int c = n0 + s * 16 + cn;
      #pragma unroll
      for (int r = 0; r < 8; ++r) {
        const int m  = m0 + ms * 16 + half8 + r;
        const int b  = m / NSEQ;
        const int ns = m % NSEQ;
        const float v = acc[r] * alpha;
        if (mode == MODE_Q || mode == MODE_K) {
          const int h = c / HD, d = c % HD;
          oh[((size_t)((b * HCNT + h) * NSEQ + ns)) * KPAD + d] = (_Float16)v;
        } else if (mode == MODE_V) {
          const int h = c / HD, d = c % HD;   // transposed store: (b,h,d,n)
          oh[((size_t)((b * HCNT + h) * HD + d)) * NSEQ + ns] = (_Float16)v;
        } else {
          of[(size_t)m * CDIM + c] = v + bias[c];
        }
      }
    }
  }
}

// ---------------------------------------------------------------------------
// Fused scores: S = (Q*scale) K^T via WMMA with the K panel streamed into LDS
// by the Tensor Data Mover (double-buffered 96-row chunks, s_wait_tensorcnt);
// fp32 row buffer in LDS; content softmax + analytic positional softmax;
// sigmoid-gated combine -> attn f16. One wave per block.
// ---------------------------------------------------------------------------
__global__ void __launch_bounds__(32, 1)
scores_softmax(const _Float16* __restrict__ qh, const _Float16* __restrict__ kh,
               const float* __restrict__ Wpos, const float* __restrict__ bpos,
               const float* __restrict__ gating,
               _Float16* __restrict__ attn, int bArg) {
  __shared__ float    srow[16 * NSEQ];           // 36,864 B
  __shared__ _Float16 kbuf[2][CHUNKR * KPAD];    // 24,576 B
  const int h  = blockIdx.y;
  const int b  = (bArg >= 0) ? bArg : (int)blockIdx.z;
  _Float16* abase = attn + ((bArg >= 0) ? 0 : (size_t)b * HCNT * NSEQ * NSEQ);
  const int n0 = blockIdx.x * 16;
  const int lane = threadIdx.x & 31;

  const _Float16* qp = qh + (size_t)((b * HCNT + h) * NSEQ) * KPAD;
  const _Float16* kp = kh + (size_t)((b * HCNT + h) * NSEQ) * KPAD;

  const v16h aq0 = load_frag_f16(qp, KPAD, n0, 0);
  const v16h aq1 = load_frag_f16(qp, KPAD, n0, 32);
  const int half8 = (lane >> 4) << 3;
  const int cn = lane & 15;

#if __has_builtin(__builtin_amdgcn_tensor_load_to_lds)
  // prologue: start DMA for chunk 0
  tdm_load_2d_f16((unsigned)(uintptr_t)&kbuf[0][0], kp,
                  KPAD, CHUNKR, KPAD, CHUNKR, KPAD);
  for (int c = 0; c < NSEQ / CHUNKR; ++c) {     // 6 chunks of 96 key rows
    if (c + 1 < NSEQ / CHUNKR) {
      tdm_load_2d_f16((unsigned)(uintptr_t)&kbuf[(c + 1) & 1][0],
                      kp + (size_t)(c + 1) * CHUNKR * KPAD,
                      KPAD, CHUNKR, KPAD, CHUNKR, KPAD);
      wait_tensorcnt<1>();   // in-order completion: chunk c is resident
    } else {
      wait_tensorcnt<0>();
    }
    const _Float16* kb = &kbuf[c & 1][0];
    for (int jj = 0; jj < CHUNKR / 16; ++jj) {
      const int j = c * (CHUNKR / 16) + jj;
      v16h bk0 = load_frag_f16(kb, KPAD, jj * 16, 0);
      v16h bk1 = load_frag_f16(kb, KPAD, jj * 16, 32);
      v8f acc = {};
      acc = WMMA_F16(aq0, bk0, acc);
      acc = WMMA_F16(aq1, bk1, acc);
      #pragma unroll
      for (int r = 0; r < 8; ++r)
        srow[(half8 + r) * NSEQ + j * 16 + cn] = acc[r];
    }
  }
#else
  for (int c = 0; c < NSEQ / CHUNKR; ++c) {
    const _Float16* ksrc = kp + (size_t)c * CHUNKR * KPAD;
    for (int i = lane; i < CHUNKR * KPAD / 8; i += 32)
      reinterpret_cast<uint4*>(&kbuf[0][0])[i] = reinterpret_cast<const uint4*>(ksrc)[i];
    __syncthreads();
    for (int jj = 0; jj < CHUNKR / 16; ++jj) {
      const int j = c * (CHUNKR / 16) + jj;
      v16h bk0 = load_frag_f16(&kbuf[0][0], KPAD, jj * 16, 0);
      v16h bk1 = load_frag_f16(&kbuf[0][0], KPAD, jj * 16, 32);
      v8f acc = {};
      acc = WMMA_F16(aq0, bk0, acc);
      acc = WMMA_F16(aq1, bk1, acc);
      #pragma unroll
      for (int r = 0; r < 8; ++r)
        srow[(half8 + r) * NSEQ + j * 16 + cn] = acc[r];
    }
    __syncthreads();
  }
#endif
  __syncthreads();

  // Softmax: 2 lanes per row, each covering 288 keys; reduce with shfl_xor.
  const int row  = lane >> 1;
  const int mbeg = (lane & 1) * (NSEQ / 2);
  const int mend = mbeg + NSEQ / 2;
  const int n  = n0 + row;
  const int qr = n / GRIDS, qc = n % GRIDS;
  const float w0 = Wpos[h * 3 + 0], w1 = Wpos[h * 3 + 1], w2 = Wpos[h * 3 + 2];
  const float bp = bpos[h];

  float smax = -1e30f, pmax = -1e30f;
  for (int m = mbeg; m < mend; ++m) {
    smax = fmaxf(smax, srow[row * NSEQ + m]);
    float dx = (float)(m % GRIDS - qc);
    float dy = (float)(m / GRIDS - qr);
    float pl = w0 * dx + w1 * dy + w2 * (dx * dx + dy * dy) + bp;
    pmax = fmaxf(pmax, pl);
  }
  smax = fmaxf(smax, __shfl_xor(smax, 1));
  pmax = fmaxf(pmax, __shfl_xor(pmax, 1));

  float ssum = 0.f, psum = 0.f;
  for (int m = mbeg; m < mend; ++m) {
    ssum += __expf(srow[row * NSEQ + m] - smax);
    float dx = (float)(m % GRIDS - qc);
    float dy = (float)(m / GRIDS - qr);
    float pl = w0 * dx + w1 * dy + w2 * (dx * dx + dy * dy) + bp;
    psum += __expf(pl - pmax);
  }
  ssum += __shfl_xor(ssum, 1);
  psum += __shfl_xor(psum, 1);

  // gated combine; both softmaxes sum to 1 so the renormalizer is exactly 1.
  const float g  = 1.f / (1.f + __expf(-gating[h]));
  const float cg = (1.f - g) / ssum;
  const float pg = g / psum;

  _Float16* arow = abase + ((size_t)h * NSEQ + n) * NSEQ;
  for (int m = mbeg; m < mend; ++m) {
    float dx = (float)(m % GRIDS - qc);
    float dy = (float)(m / GRIDS - qr);
    float pl = w0 * dx + w1 * dy + w2 * (dx * dx + dy * dy) + bp;
    float val = cg * __expf(srow[row * NSEQ + m] - smax) + pg * __expf(pl - pmax);
    arow[m] = (_Float16)val;
  }
}

// ---------------------------------------------------------------------------
// out_heads(16 x 48 head tile per wave) = attn @ V (V pre-transposed), K=576.
// K loop unrolled x2 with ping-pong fragment sets (no register copies).
// ---------------------------------------------------------------------------
__global__ void __launch_bounds__(32, 1)
attn_v_wmma(const _Float16* __restrict__ attn, const _Float16* __restrict__ vT,
            _Float16* __restrict__ outh, int bArg) {
  const int h  = blockIdx.y;
  const int b  = (bArg >= 0) ? bArg : (int)blockIdx.z;
  const int n0 = blockIdx.x * 16;
  const int lane = threadIdx.x & 31;

  const _Float16* ap = attn + ((bArg >= 0) ? 0 : (size_t)b * HCNT * NSEQ * NSEQ)
                            + (size_t)h * NSEQ * NSEQ;
  const _Float16* vp = vT + ((size_t)((b * HCNT + h) * HD)) * NSEQ;

  v8f acc0 = {}, acc1 = {}, acc2 = {};
  v16h aA, vA0, vA1, vA2;   // ping set
  v16h aB, vB0, vB1, vB2;   // pong set

#define AV_LOADSET(SUF, KK)                               \
  a##SUF  = load_frag_f16(ap, NSEQ, n0, (KK));            \
  v##SUF##0 = load_frag_f16(vp, NSEQ,  0, (KK));          \
  v##SUF##1 = load_frag_f16(vp, NSEQ, 16, (KK));          \
  v##SUF##2 = load_frag_f16(vp, NSEQ, 32, (KK));

#define AV_WMMASET(SUF)                                   \
  acc0 = WMMA_F16(a##SUF, v##SUF##0, acc0);               \
  acc1 = WMMA_F16(a##SUF, v##SUF##1, acc1);               \
  acc2 = WMMA_F16(a##SUF, v##SUF##2, acc2);

  AV_LOADSET(A, 0)
  for (int k0 = 0; k0 < NSEQ - 64; k0 += 64) {   // 8 iterations (18 K-steps total)
    AV_LOADSET(B, k0 + 32)
    AV_WMMASET(A)
    AV_LOADSET(A, k0 + 64)
    AV_WMMASET(B)
  }
  AV_LOADSET(B, NSEQ - 32)
  AV_WMMASET(A)
  AV_WMMASET(B)
#undef AV_LOADSET
#undef AV_WMMASET

  const int half8 = (lane >> 4) << 3;
  const int cn = lane & 15;
  #pragma unroll
  for (int r = 0; r < 8; ++r) {
    const int n = n0 + half8 + r;
    _Float16* orow = outh + ((size_t)(b * NSEQ + n)) * CDIM + h * HD;
    orow[cn]      = (_Float16)acc0[r];
    orow[16 + cn] = (_Float16)acc1[r];
    orow[32 + cn] = (_Float16)acc2[r];
  }
}

// ---------------------------------------------------------------------------
// Host-side orchestration
// ---------------------------------------------------------------------------
extern "C" void kernel_launch(void* const* d_in, const int* in_sizes, int n_in,
                              void* d_out, int out_size, void* d_ws, size_t ws_size,
                              hipStream_t stream) {
  (void)in_sizes; (void)n_in; (void)out_size;
  const float* x      = (const float*)d_in[0];
  const float* Wq     = (const float*)d_in[1];
  const float* Wk     = (const float*)d_in[2];
  const float* Wv     = (const float*)d_in[3];
  const float* Wproj  = (const float*)d_in[4];
  const float* bproj  = (const float*)d_in[5];
  const float* Wpos   = (const float*)d_in[6];
  const float* bpos   = (const float*)d_in[7];
  const float* gating = (const float*)d_in[8];
  float* out = (float*)d_out;

  char* wsp = (char*)d_ws;
  auto carve = [&](size_t bytes) {
    char* p = wsp;
    wsp += (bytes + 255) & ~(size_t)255;
    return p;
  };
  const size_t SZ_XH   = (size_t)MROWS * CDIM;
  const size_t SZ_W    = (size_t)CDIM * CDIM;
  const size_t SZ_QK   = (size_t)BATCH * HCNT * NSEQ * KPAD;
  const size_t SZ_VT   = (size_t)BATCH * HCNT * HD * NSEQ;
  const size_t SZ_ATTN = (size_t)HCNT * NSEQ * NSEQ;      // per-batch

  _Float16* xh   = (_Float16*)carve(SZ_XH * 2);
  _Float16* Wqh  = (_Float16*)carve(SZ_W * 2);
  _Float16* Wkh  = (_Float16*)carve(SZ_W * 2);
  _Float16* Wvh  = (_Float16*)carve(SZ_W * 2);
  _Float16* Wph  = (_Float16*)carve(SZ_W * 2);
  _Float16* qh   = (_Float16*)carve(SZ_QK * 2);
  _Float16* kh   = (_Float16*)carve(SZ_QK * 2);
  _Float16* vT   = (_Float16*)carve(SZ_VT * 2);
  _Float16* outh = (_Float16*)carve(SZ_XH * 2);
  const size_t fixed = (size_t)(wsp - (char*)d_ws);
  // batched attention buffer if the workspace allows (decision depends only
  // on ws_size -> deterministic)
  const bool batched = ws_size >= fixed + SZ_ATTN * 2 * BATCH + 4096;
  _Float16* attn = (_Float16*)carve(SZ_ATTN * 2 * (batched ? BATCH : 1));

  auto cvt = [&](const float* s, _Float16* d, size_t n) {
    cvt_f32_to_f16<<<dim3((unsigned)((n + 255) / 256)), dim3(256), 0, stream>>>(s, d, (int)n);
  };
  cvt(x, xh, SZ_XH);
  cvt(Wq, Wqh, SZ_W);
  cvt(Wk, Wkh, SZ_W);
  cvt(Wv, Wvh, SZ_W);
  cvt(Wproj, Wph, SZ_W);

  hipMemsetAsync(qh, 0, SZ_QK * 2, stream);   // zero K-pad columns 48..63
  hipMemsetAsync(kh, 0, SZ_QK * 2, stream);

  const dim3 gemmGrid(216), gemmBlk(256);     // 216*8 waves = 144*12 wave-tiles
  const float scale = 0.14433756729740643f;   // 48^-0.5, folded into Q
  gemm768_wmma<<<gemmGrid, gemmBlk, 0, stream>>>(xh, Wqh, qh, nullptr, nullptr, MODE_Q, scale);
  gemm768_wmma<<<gemmGrid, gemmBlk, 0, stream>>>(xh, Wkh, kh, nullptr, nullptr, MODE_K, 1.0f);
  gemm768_wmma<<<gemmGrid, gemmBlk, 0, stream>>>(xh, Wvh, vT, nullptr, nullptr, MODE_V, 1.0f);

  if (batched) {
    scores_softmax<<<dim3(NSEQ / 16, HCNT, BATCH), dim3(32), 0, stream>>>(
        qh, kh, Wpos, bpos, gating, attn, -1);
    attn_v_wmma<<<dim3(NSEQ / 16, HCNT, BATCH), dim3(32), 0, stream>>>(
        attn, vT, outh, -1);
  } else {
    for (int b = 0; b < BATCH; ++b) {
      scores_softmax<<<dim3(NSEQ / 16, HCNT), dim3(32), 0, stream>>>(
          qh, kh, Wpos, bpos, gating, attn, b);
      attn_v_wmma<<<dim3(NSEQ / 16, HCNT), dim3(32), 0, stream>>>(
          attn, vT, outh, b);
    }
  }

  gemm768_wmma<<<gemmGrid, gemmBlk, 0, stream>>>(outh, Wph, nullptr, out, bproj, MODE_PROJ, 1.0f);
}
